// LSTM_noemb_76785425318296
// MI455X (gfx1250) — matrix operand order repaired and checked by reading, not compiled, not run on hardware
//
#include <hip/hip_runtime.h>

typedef __attribute__((ext_vector_type(16))) _Float16 v16h;
typedef __attribute__((ext_vector_type(8)))  _Float16 v8h;
typedef __attribute__((ext_vector_type(8)))  float    v8f;
typedef int v4i_ __attribute__((vector_size(16)));   // int4, matches builtin signature
typedef __attribute__((address_space(1))) v4i_* gptr_b128;  // global int4*
typedef __attribute__((address_space(3))) v4i_* lptr_b128;  // LDS int4*

namespace {
constexpr int kH       = 512;     // hidden
constexpr int kG       = 2048;    // 4*H gate columns
constexpr int kB       = 64;      // batch
constexpr int kT       = 64;      // timesteps
constexpr int kV       = 50257;   // vocab
constexpr int kNWG     = 16;      // workgroups per branch
constexpr int kThreads = 256;     // 8 waves (wave32)
constexpr int kSlab    = 32;      // hidden units per WG -> 128 gate cols
constexpr int kAStr    = kH + 8;  // LDS activation row stride (f16), bank-padded
constexpr int kGStr    = 132;     // LDS gate row stride (f32), bank-padded
}

struct LstmParams {
  const int*      xidx[2];
  const float*    wih0[2];
  const float*    bih0[2];
  const float*    bhh0[2];
  const float*    bih1[2];
  const float*    bhh1[2];
  const _Float16* whhT0[2];   // (512 x 2048) f16, K-major
  const _Float16* whhT1[2];
  const _Float16* wih1T[2];
  _Float16*       out0[2];    // layer-0 output, [t][b][h] f16
  _Float16*       hbuf[2];    // double-buffered h, [2][b][h] f16
  int*            cnt;        // per-branch arrive counters
  float*          out;        // (2,B,T,H) f32
};

__device__ __forceinline__ float sigm(float x) { return 1.0f / (1.0f + __expf(-x)); }

// ---- gfx1250 async global->LDS copy (ASYNCcnt-tracked, bypasses VGPRs) ----
__device__ __forceinline__ void async_copy_b128(const void* g, void* l)
{
#if __has_builtin(__builtin_amdgcn_global_load_async_to_lds_b128)
  __builtin_amdgcn_global_load_async_to_lds_b128(
      (gptr_b128)(void*)g, (lptr_b128)l, 0, 0);
#else
  unsigned loff = (unsigned)(size_t)(__attribute__((address_space(3))) char*)l;
  asm volatile("global_load_async_to_lds_b128 %0, %1, off"
               :: "v"(loff), "v"(g) : "memory");
#endif
}

__device__ __forceinline__ void wait_async0()
{
#if __has_builtin(__builtin_amdgcn_s_wait_asynccnt)
  __builtin_amdgcn_s_wait_asynccnt(0);
#else
  asm volatile("s_wait_asynccnt 0" ::: "memory");
#endif
}

// Cooperative 64x512 f16 tile fill into padded LDS via async DMA.
__device__ __forceinline__ void fill_tile_async(_Float16 (*sA)[kAStr],
                                                const _Float16* src, int tid)
{
  for (int i = tid; i < kB * (kH / 8); i += kThreads) {
    int r = i >> 6, c8 = (i & 63) * 8;
    async_copy_b128(&src[r * kH + c8], &sA[r][c8]);
  }
  wait_async0();
}

// A fragment (16x32 f16, M x K) from LDS activation tile, per ISA 16-bit A layout:
// lanes 0-15: M=lane, halves 0-7 = K 0..7,  halves 8-15 = K 16..23
// lanes 16-31: M=lane-16, halves 0-7 = K 8..15, halves 8-15 = K 24..31
__device__ __forceinline__ v16h load_a_frag(const _Float16* sA, int lane, int mtile, int kb)
{
  int row  = mtile * 16 + (lane & 15);
  int koff = (lane >> 4) * 8;
  const _Float16* p = sA + row * kAStr + kb + koff;
  union { v16h v; v8h h[2]; } u;
  u.h[0] = *(const v8h*)(p);
  u.h[1] = *(const v8h*)(p + 16);
  return u.v;
}

// B fragment (32x16 f16, K x N) from K-major transposed weights in global (L2-resident):
// N = lane%16; lanes 0-15: halves 0-7 = K 0..7, 8-15 = K 16..23; lanes 16-31: K 8..15, K 24..31.
__device__ __forceinline__ v16h load_b_frag(const _Float16* wT, int lane, int kb, int gcol)
{
  const _Float16* p = wT + (size_t)(kb + (lane >> 4) * 8) * kG + gcol + (lane & 15);
  v16h v;
  #pragma unroll
  for (int s = 0; s < 8; ++s) {
    v[s]     = p[(size_t)s * kG];
    v[s + 8] = p[(size_t)(s + 16) * kG];
  }
  return v;
}

// One (64 x 128) += (64 x 512) x (512 x 128) pass: 512 WMMAs / WG, 64 per wave.
__device__ __forceinline__ void gemm_pass(v8f acc[4], const _Float16* sA, const _Float16* wT,
                                          int lane, int mtile, int jbase, int u0)
{
  #pragma unroll 4
  for (int kt = 0; kt < 16; ++kt) {
    int kb = kt * 32;
    v16h a = load_a_frag(sA, lane, mtile, kb);
    #pragma unroll
    for (int jj = 0; jj < 4; ++jj) {
      int j    = jbase + jj;
      int gcol = (j >> 1) * kH + u0 + (j & 1) * 16;   // gate-major global column
      if (jj == 0 && kt + 1 < 16)                      // gfx1250 global_prefetch_b8
        __builtin_prefetch(wT + (size_t)(kb + 32 + lane) * kG + gcol, 0, 3);
      v16h b = load_b_frag(wT, lane, kb, gcol);
      acc[jj] = __builtin_amdgcn_wmma_f32_16x16x32_f16(
          false, a, false, b, (short)0, acc[jj], false, false);
    }
  }
}

__global__ void __launch_bounds__(kThreads, 1)
lstm_persistent(LstmParams P)
{
  __shared__ __align__(16) _Float16 sA[kB][kAStr];   // 66.5 KB activation tile (h or x)
  __shared__ __align__(16) float    sG[kB][kGStr];   // 33.8 KB gate staging
  __shared__ __align__(16) float    sC[kB][kSlab];   //  8.2 KB cell state slab

  const int wg     = blockIdx.x;
  const int branch = wg >> 4;
  const int slab   = wg & 15;
  const int u0     = slab * kSlab;
  const int tid    = threadIdx.x;
  const int lane   = tid & 31;
  const int wave   = tid >> 5;
  const int mtile  = wave & 3;          // batch tile (4 x 16 = 64)
  const int jbase  = (wave >> 2) * 4;   // 4 N-tiles per wave (8 total)

  int*         cnt  = P.cnt + branch;
  const int*   xq   = P.xidx[branch];
  const float* wih0 = P.wih0[branch];
  _Float16*    out0 = P.out0[branch];
  _Float16*    hb   = P.hbuf[branch];
  float*       dout = P.out + (size_t)branch * kB * kT * kH;

  int gstep = 0;
  for (int layer = 0; layer < 2; ++layer) {
    const _Float16* whhT = layer ? P.whhT1[branch] : P.whhT0[branch];
    const _Float16* wxT  = P.wih1T[branch];
    const float*    bi   = layer ? P.bih1[branch] : P.bih0[branch];
    const float*    bh   = layer ? P.bhh1[branch] : P.bhh0[branch];

    for (int t = 0; t < kT; ++t, ++gstep) {
      // ---- grid barrier: wait until every WG of this branch finished prior step ----
      if (gstep > 0) {
        if (tid == 0) {
          while (__hip_atomic_load(cnt, __ATOMIC_ACQUIRE, __HIP_MEMORY_SCOPE_AGENT)
                 < gstep * kNWG)
            __builtin_amdgcn_s_sleep(2);
        }
        __syncthreads();
      }

      v8f zero = {0.f, 0.f, 0.f, 0.f, 0.f, 0.f, 0.f, 0.f};
      v8f acc[4] = {zero, zero, zero, zero};

      const bool do_h = (t > 0);   // h0 == 0: skip recurrent pass entirely at t==0
      if (do_h) {
        fill_tile_async(sA, hb + (size_t)(t & 1) * kB * kH, tid);
        __syncthreads();
        gemm_pass(acc, &sA[0][0], whhT, lane, mtile, jbase, u0);
        __syncthreads();
      }

      if (layer == 1) {            // layer-1 input projection, fused per-timestep
        fill_tile_async(sA, out0 + (size_t)t * kB * kH, tid);
        __syncthreads();
        gemm_pass(acc, &sA[0][0], wxT, lane, mtile, jbase, u0);
        __syncthreads();
      }

      // ---- scatter accumulators to LDS gate buffer (C layout: vgpr r, lane n) ----
      const bool haveG = do_h || (layer == 1);
      if (haveG) {
        int radd = (lane >> 4) << 3;
        int ncol = lane & 15;
        #pragma unroll
        for (int jj = 0; jj < 4; ++jj)
          #pragma unroll
          for (int r = 0; r < 8; ++r)
            sG[mtile * 16 + r + radd][(jbase + jj) * 16 + ncol] = acc[jj][r];
      }
      __syncthreads();

      // ---- elementwise LSTM cell for this WG's 32 hidden units x 64 batches ----
      #pragma unroll 2
      for (int it = 0; it < (kB * kSlab) / kThreads; ++it) {
        int item = it * kThreads + tid;
        int b  = item >> 5;
        int uu = item & 31;
        int ci = u0 + uu;
        float gi = haveG ? sG[b][uu]      : 0.f;
        float gf = haveG ? sG[b][32 + uu] : 0.f;
        float gg = haveG ? sG[b][64 + uu] : 0.f;
        float go = haveG ? sG[b][96 + uu] : 0.f;
        if (layer == 0) {            // one-hot matmul == row gather of Wih0
          size_t tok = (size_t)(unsigned)xq[b * kT + t];
          gi += wih0[(size_t)(0 * kH + ci) * kV + tok];
          gf += wih0[(size_t)(1 * kH + ci) * kV + tok];
          gg += wih0[(size_t)(2 * kH + ci) * kV + tok];
          go += wih0[(size_t)(3 * kH + ci) * kV + tok];
        }
        gi += bi[0 * kH + ci] + bh[0 * kH + ci];
        gf += bi[1 * kH + ci] + bh[1 * kH + ci];
        gg += bi[2 * kH + ci] + bh[2 * kH + ci];
        go += bi[3 * kH + ci] + bh[3 * kH + ci];
        float iv = sigm(gi), fv = sigm(gf), gv = tanhf(gg), ov = sigm(go);
        float cp = (t == 0) ? 0.f : sC[b][uu];
        float cv = fv * cp + iv * gv;
        float hv = ov * tanhf(cv);
        sC[b][uu] = cv;
        hb[(size_t)((t + 1) & 1) * kB * kH + b * kH + ci] = (_Float16)hv;
        if (layer == 0)
          out0[(size_t)t * kB * kH + b * kH + ci] = (_Float16)hv;
        else
          dout[((size_t)b * kT + t) * kH + ci] = hv;
      }

      // ---- publish h, arrive ----
      __threadfence();
      __syncthreads();
      if (tid == 0)
        __hip_atomic_fetch_add(cnt, 1, __ATOMIC_RELEASE, __HIP_MEMORY_SCOPE_AGENT);
    }
  }
}

// Prep: transpose (2048 x 512) f32 row-major -> (512 x 2048) f16 K-major.
__global__ void transpose_to_f16(const float* __restrict__ src, _Float16* __restrict__ dst)
{
  int i = blockIdx.x * blockDim.x + threadIdx.x;
  if (i < kG * kH) {
    int n = i / kH;
    int k = i % kH;
    dst[(size_t)k * kG + n] = (_Float16)src[i];
  }
}

__global__ void zero_counters(int* cnt)
{
  if (threadIdx.x < 2) cnt[threadIdx.x] = 0;
}

extern "C" void kernel_launch(void* const* d_in, const int* in_sizes, int n_in,
                              void* d_out, int out_size, void* d_ws, size_t ws_size,
                              hipStream_t stream)
{
  (void)in_sizes; (void)n_in; (void)out_size; (void)ws_size;

  // Input order: x_q, x_f, then per branch (q,f) per layer (0,1): Wih, Whh, bih, bhh
  const int kW0[2] = {2, 10};   // Wih_*0
  const int kH0[2] = {3, 11};   // Whh_*0
  const int kBi0[2] = {4, 12}, kBh0[2] = {5, 13};
  const int kW1[2] = {6, 14};   // Wih_*1
  const int kH1[2] = {7, 15};   // Whh_*1
  const int kBi1[2] = {8, 16}, kBh1[2] = {9, 17};

  char* ws = (char*)d_ws;
  size_t off = 0;
  auto carve = [&](size_t bytes) -> void* {
    void* p = ws + off;
    off += (bytes + 255) & ~(size_t)255;
    return p;
  };

  int* cnt = (int*)carve(256);
  LstmParams P;
  for (int br = 0; br < 2; ++br) {
    P.whhT0[br] = (_Float16*)carve((size_t)kH * kG * sizeof(_Float16));
    P.whhT1[br] = (_Float16*)carve((size_t)kH * kG * sizeof(_Float16));
    P.wih1T[br] = (_Float16*)carve((size_t)kH * kG * sizeof(_Float16));
    P.out0[br]  = (_Float16*)carve((size_t)kT * kB * kH * sizeof(_Float16));
    P.hbuf[br]  = (_Float16*)carve((size_t)2 * kB * kH * sizeof(_Float16));
  }

  zero_counters<<<1, 32, 0, stream>>>(cnt);

  const int tb = 256;
  const int tg = (kG * kH + tb - 1) / tb;
  for (int br = 0; br < 2; ++br) {
    transpose_to_f16<<<tg, tb, 0, stream>>>((const float*)d_in[kH0[br]],
                                            (_Float16*)P.whhT0[br]);
    transpose_to_f16<<<tg, tb, 0, stream>>>((const float*)d_in[kH1[br]],
                                            (_Float16*)P.whhT1[br]);
    transpose_to_f16<<<tg, tb, 0, stream>>>((const float*)d_in[kW1[br]],
                                            (_Float16*)P.wih1T[br]);
  }

  P.xidx[0] = (const int*)d_in[0];
  P.xidx[1] = (const int*)d_in[1];
  for (int br = 0; br < 2; ++br) {
    P.wih0[br] = (const float*)d_in[kW0[br]];
    P.bih0[br] = (const float*)d_in[kBi0[br]];
    P.bhh0[br] = (const float*)d_in[kBh0[br]];
    P.bih1[br] = (const float*)d_in[kBi1[br]];
    P.bhh1[br] = (const float*)d_in[kBh1[br]];
  }
  P.cnt = cnt;
  P.out = (float*)d_out;

  lstm_persistent<<<2 * kNWG, kThreads, 0, stream>>>(P);
}
